// ParserModel_57492432224895
// MI455X (gfx1250) — compile-verified, matrix operand order checked by reading, not run on hardware
//
#include <hip/hip_runtime.h>
#include <cstdint>
#include <cstddef>

typedef __attribute__((ext_vector_type(16))) __bf16 v16bf;
typedef __attribute__((ext_vector_type(8)))  float  v8f;
typedef uint32_t __attribute__((ext_vector_type(4))) v4u;
typedef int      __attribute__((ext_vector_type(4))) v4i;
typedef int      __attribute__((ext_vector_type(8))) v8i;
typedef unsigned short u16;

constexpr int BATCH = 16384;
constexpr int EMB   = 128;
constexpr int KTOT  = 6144;   // 48 * 128
constexpr int HID   = 1024;
constexpr int NWF = 18, NTF = 18, NDF = 12, NFEAT = 48;

constexpr int BM = 128, BN = 128;
constexpr int LDH = 136;      // LDS row stride in halves (272 B = 68 DW = 64 + 4 pad)

// TDM availability (device pass on gfx1250 toolchains only; host pass falls back)
#if defined(__has_builtin)
#  if __has_builtin(__builtin_amdgcn_tensor_load_to_lds) && __has_builtin(__builtin_amdgcn_s_wait_tensorcnt)
#    define USE_TDM 1
#  endif
#endif
#ifndef USE_TDM
#  define USE_TDM 0
#endif

__device__ __forceinline__ u16 f32_to_bf16(float f) {
    uint32_t x = __float_as_uint(f);
    uint32_t r = x + 0x7FFFu + ((x >> 16) & 1u);   // RNE (off the hot path)
    return (u16)(r >> 16);
}

__device__ __forceinline__ float bf16_to_f32(u16 h) {
    return __uint_as_float(((uint32_t)h) << 16);
}

// pack trunc-bf16(lo), trunc-bf16(hi) into one dword via v_perm_b32
__device__ __forceinline__ uint32_t pack_bf16(float lo, float hi) {
    return __builtin_amdgcn_perm(__float_as_uint(hi), __float_as_uint(lo), 0x07060302u);
}

union Frag {
    struct { uint4 lo, hi; } u;
    v16bf v;
};

#if USE_TDM
// Issue one TDM 2D tile load: Wt tile (128 rows x 128 bf16, row stride KTOT)
// -> LDS at ldsByteAddr with +4 DWORD padding after every 64 DWORDs (LDH=136).
__device__ __forceinline__ void tdm_load_b_tile(const u16* gptr, uint32_t ldsByteAddr) {
    const uint64_t ga = (uint64_t)(uintptr_t)gptr;
    v4u g0;
    g0.x = 1u;                                   // count=1, user descriptor
    g0.y = ldsByteAddr;                          // lds_addr
    g0.z = (uint32_t)ga;                         // global_addr[31:0]
    g0.w = (uint32_t)(ga >> 32) | (2u << 30);    // global_addr[56:32] | type=2

    v8i g1;
    // workgroup_mask=0 | data_size=1(2B)<<16 | pad_enable<<20 |
    // pad_interval=5(64DW)<<22 | pad_amount=3(4DW)<<25
    g1[0] = (int)((1u << 16) | (1u << 20) | (5u << 22) | (3u << 25));
    g1[1] = (int)(((uint32_t)KTOT & 0xFFFFu) << 16);                 // tensor_dim0 lo16
    g1[2] = (int)((((uint32_t)KTOT >> 16) & 0xFFFFu) |
                  (((uint32_t)HID & 0xFFFFu) << 16));                // dim0 hi16 | dim1 lo16
    g1[3] = (int)((((uint32_t)HID >> 16) & 0xFFFFu) | (128u << 16)); // dim1 hi16 | tile_dim0=128
    g1[4] = (int)(128u);                                             // tile_dim1=128, tile_dim2=0
    g1[5] = (int)(uint32_t)KTOT;                                     // tensor_dim0_stride lo32
    g1[6] = 0;                                                       // stride hi16 | dim1_stride lo16
    g1[7] = 0;

    v4i z4 = (v4i){};
#  if __has_include(<hip/amd_detail/amd_gfx1250_TDM.h>)
    v8i z8 = (v8i){};
    __builtin_amdgcn_tensor_load_to_lds(g0, g1, z4, z4, z8, 0);      // clang-23 arity
#  else
    __builtin_amdgcn_tensor_load_to_lds(g0, g1, z4, z4, 0);          // ROCm 7.2 arity
#  endif
}
#endif

// ---------------------------------------------------------------------------
// Kernel 1: W_h (6144x1024 f32, K-major) -> Wt (1024x6144 bf16, N-major)
// ---------------------------------------------------------------------------
__global__ __launch_bounds__(256)
void convert_wh(const float* __restrict__ Wh, u16* __restrict__ Wt) {
    int idx = blockIdx.x * 256 + threadIdx.x;      // 0 .. 6144*1024-1
    int k = idx / HID;
    int n = idx % HID;
    Wt[(size_t)n * KTOT + k] = f32_to_bf16(Wh[idx]);
}

// ---------------------------------------------------------------------------
// Kernel 2: fused gather + GEMM1 + bias + ReLU  ->  h (16384x1024 bf16)
//   grid = (BATCH/BM, HID/BN), block = 256 (8 waves, 2x4 wave grid)
//   outer loop over 48 features; B tile streamed by TDM, A tile gathered+packed
// ---------------------------------------------------------------------------
__global__ __launch_bounds__(256)
void fused_gemm1(const int*   __restrict__ wid,
                 const int*   __restrict__ tgid,
                 const int*   __restrict__ did,
                 const float* __restrict__ wemb,
                 const float* __restrict__ temb,
                 const float* __restrict__ demb,
                 const u16*   __restrict__ Wt,
                 const float* __restrict__ bh,
                 u16*         __restrict__ h)
{
    __shared__ u16 As[BM * LDH];   // 34816 B
    __shared__ u16 Bs[BN * LDH];   // 34816 B

    const int tid  = threadIdx.x;
    const int lane = tid & 31;
    const int wave = tid >> 5;
    const int wm   = wave >> 2;           // 0..1  -> 64 rows each
    const int wn   = wave & 3;            // 0..3  -> 32 cols each
    const int m0   = blockIdx.x * BM;
    const int n0   = blockIdx.y * BN;

    v8f acc[4][2];
    #pragma unroll
    for (int i = 0; i < 4; i++)
        #pragma unroll
        for (int j = 0; j < 2; j++)
            acc[i][j] = (v8f){};

    const int halfSel = (lane < 16) ? 0 : 8;   // K {0..7,16..23} vs {8..15,24..31}
    const int r = lane & 15;

#if USE_TDM
    const uint32_t bsLds = (uint32_t)(uintptr_t)(&Bs[0]);   // LDS byte address of Bs
#endif

    for (int f = 0; f < NFEAT; ++f) {
        // ---- B tile: TDM DMA (or vector-copy fallback) ----
#if USE_TDM
        if (wave == 0) {
            tdm_load_b_tile(Wt + (size_t)n0 * KTOT + f * EMB, bsLds);
        }
#else
        #pragma unroll
        for (int s = 0; s < 2; ++s) {
            const int slot = tid + s * 256;
            const int row  = slot >> 2;
            const int seg  = (slot & 3) * 32;
            const uint4* src = (const uint4*)(Wt + (size_t)(n0 + row) * KTOT + f * EMB + seg);
            uint4* dst = (uint4*)&Bs[row * LDH + seg];
            #pragma unroll
            for (int u = 0; u < 4; ++u) dst[u] = src[u];
        }
#endif

        // ---- A tile: gather one full 128-wide embedding row per M row ----
        // 512 slots (row, quarter), 2 per thread; 32 f32 -> 16 packed dwords each
        #pragma unroll
        for (int s = 0; s < 2; ++s) {
            const int slot = tid + s * 256;
            const int row  = slot >> 2;          // 0..127
            const int seg  = (slot & 3) * 32;    // half offset 0/32/64/96
            const int gm   = m0 + row;
            int id; const float* tab;
            if (f < NWF)            { id = wid [gm * NWF + f];               tab = wemb; }
            else if (f < NWF + NTF) { id = tgid[gm * NTF + (f - NWF)];       tab = temb; }
            else                    { id = did [gm * NDF + (f - NWF - NTF)]; tab = demb; }
            const float4* src = (const float4*)(tab + (size_t)id * EMB + seg);
            uint4* dst = (uint4*)&As[row * LDH + seg];
            #pragma unroll
            for (int u = 0; u < 4; ++u) {
                float4 a = src[2 * u];
                float4 b = src[2 * u + 1];
                uint4 o;
                o.x = pack_bf16(a.x, a.y);
                o.y = pack_bf16(a.z, a.w);
                o.z = pack_bf16(b.x, b.y);
                o.w = pack_bf16(b.z, b.w);
                dst[u] = o;
            }
        }

#if USE_TDM
        if (wave == 0) {
            __builtin_amdgcn_s_wait_tensorcnt(0);  // DMA landed in LDS
        }
#endif
        __syncthreads();

        // ---- 4 K=32 chunks, 8 WMMAs each (32 static v_wmma) ----
        #pragma unroll
        for (int kk = 0; kk < 4; ++kk) {
            const int kb = kk * 32 + halfSel;
            Frag afr[4], bfr[2];
            #pragma unroll
            for (int i = 0; i < 4; i++) {
                const u16* p = &As[(wm * 64 + i * 16 + r) * LDH + kb];
                afr[i].u.lo = *(const uint4*)(p);
                afr[i].u.hi = *(const uint4*)(p + 16);
            }
            #pragma unroll
            for (int j = 0; j < 2; j++) {
                const u16* p = &Bs[(wn * 32 + j * 16 + r) * LDH + kb];
                bfr[j].u.lo = *(const uint4*)(p);
                bfr[j].u.hi = *(const uint4*)(p + 16);
            }
            #pragma unroll
            for (int i = 0; i < 4; i++)
                #pragma unroll
                for (int j = 0; j < 2; j++)
                    acc[i][j] = __builtin_amdgcn_wmma_f32_16x16x32_bf16(
                        /*neg_a=*/false, afr[i].v,
                        /*neg_b=*/false, bfr[j].v,
                        /*c_mod=*/(short)0, acc[i][j],
                        /*reuse_a=*/false, /*reuse_b=*/false);
        }
        __syncthreads();
    }

    // ---- epilogue: bias + ReLU, store bf16 h (C/D layout: VGPR v -> M, lane%16 -> N)
    const int mAdd = (lane < 16) ? 0 : 8;
    #pragma unroll
    for (int j = 0; j < 2; j++) {
        const int col  = n0 + wn * 32 + j * 16 + (lane & 15);
        const float bias = bh[col];
        #pragma unroll
        for (int i = 0; i < 4; i++) {
            #pragma unroll
            for (int v = 0; v < 8; v++) {
                const int row = m0 + wm * 64 + i * 16 + v + mAdd;
                float val = acc[i][j][v] + bias;
                val = val > 0.0f ? val : 0.0f;
                h[(size_t)row * HID + col] = f32_to_bf16(val);
            }
        }
    }
}

// ---------------------------------------------------------------------------
// Kernel 3: out = h @ W_o + b_o   (N=3, one wave per row)
// ---------------------------------------------------------------------------
__global__ __launch_bounds__(256)
void gemm2(const u16* __restrict__ h, const float* __restrict__ Wo,
           const float* __restrict__ bo, float* __restrict__ out)
{
    const int wave = threadIdx.x >> 5;
    const int lane = threadIdx.x & 31;
    const int row  = blockIdx.x * 8 + wave;

    float p0 = 0.f, p1 = 0.f, p2 = 0.f;
    for (int k = lane; k < HID; k += 32) {
        float hv = bf16_to_f32(h[(size_t)row * HID + k]);
        p0 += hv * Wo[k * 3 + 0];
        p1 += hv * Wo[k * 3 + 1];
        p2 += hv * Wo[k * 3 + 2];
    }
    #pragma unroll
    for (int m = 16; m >= 1; m >>= 1) {
        p0 += __shfl_xor(p0, m, 32);
        p1 += __shfl_xor(p1, m, 32);
        p2 += __shfl_xor(p2, m, 32);
    }
    if (lane == 0) {
        out[row * 3 + 0] = p0 + bo[0];
        out[row * 3 + 1] = p1 + bo[1];
        out[row * 3 + 2] = p2 + bo[2];
    }
}

// ---------------------------------------------------------------------------
extern "C" void kernel_launch(void* const* d_in, const int* in_sizes, int n_in,
                              void* d_out, int out_size, void* d_ws, size_t ws_size,
                              hipStream_t stream)
{
    const int*   wid  = (const int*)  d_in[0];   // (B, 18)
    const int*   tgid = (const int*)  d_in[1];   // (B, 18)
    const int*   did  = (const int*)  d_in[2];   // (B, 12)
    const float* wemb = (const float*)d_in[3];   // (100000, 128)
    const float* temb = (const float*)d_in[4];   // (48, 128)
    const float* demb = (const float*)d_in[5];   // (40, 128)
    const float* Wh   = (const float*)d_in[6];   // (6144, 1024)
    const float* bh   = (const float*)d_in[7];   // (1024,)
    const float* Wo   = (const float*)d_in[8];   // (1024, 3)
    const float* bo   = (const float*)d_in[9];   // (3,)
    float*       out  = (float*)d_out;           // (16384, 3)

    u16* Wt = (u16*)d_ws;                        // 6144*1024 bf16  (12.6 MB)
    u16* hb = Wt + (size_t)KTOT * HID;           // 16384*1024 bf16 (33.6 MB)

    convert_wh<<<(KTOT * HID) / 256, 256, 0, stream>>>(Wh, Wt);
    fused_gemm1<<<dim3(BATCH / BM, HID / BN), 256, 0, stream>>>(
        wid, tgid, did, wemb, temb, demb, Wt, bh, hb);
    gemm2<<<BATCH / 8, 256, 0, stream>>>(hb, Wo, bo, out);
}
